// microcircuit_56770877719048
// MI455X (gfx1250) — compile-verified, matrix operand order checked by reading, not run on hardware
//
#include <hip/hip_runtime.h>
#include <math.h>

// Problem constants (from reference)
#define DD   1408   // hidden D
#define FF   512    // feature F
#define BB   64     // batch
#define TT   512    // time
#define S0   128    // L4 slice (only nonzero W_ih rows)
#define S1   320    // L23 slice (output tap)

// Tile counts
#define NT_HH 88    // 1408/16  N-tiles of Whh
#define KT_HH 44    // 1408/32  K-tiles of Whh
#define LDS_ROW 1416 // padded bf16 row stride for staged h (kills bank conflicts)

#define RNN_BLOCKS 44
#define RNN_LDS_BYTES (BB * LDS_ROW * 2 + 8 * 32 * 32 * 4)  // h stage + K-reduction

typedef __attribute__((ext_vector_type(16))) __bf16 v16bf;
typedef __attribute__((ext_vector_type(8)))  __bf16 v8bf;
typedef __attribute__((ext_vector_type(8)))  float  v8f;
typedef __attribute__((ext_vector_type(4)))  float  v4f;

union BF16x16 { v16bf v; v8bf h[2]; };

// K index of element j (0..15) of a 16-bf16 A/B fragment held by `lane`
// (ISA 7.12.2: lanes 0-15 hold K={0..7,16..23}; lanes 16-31 hold K={8..15,24..31})
static __device__ __forceinline__ int kmap(int lane, int j) {
  return ((lane & 16) ? 8 : 0) + (j < 8 ? j : j + 8);
}

// ---------------------------------------------------------------------------
// Phase 0: pack weights into per-(n_tile,k_tile) B-fragment blobs:
// blob layout = [tile][lane 0..31][16 bf16], so the GEMM hot loop is two
// perfectly coalesced b128 loads per lane.
// ---------------------------------------------------------------------------
__global__ void pack_whh(const float* __restrict__ W, const unsigned char* __restrict__ M,
                         __bf16* __restrict__ out) {
  int idx = blockIdx.x * 256 + threadIdx.x;          // one packed bf16 element
  int tile = idx >> 9, within = idx & 511;
  int lane = within >> 4, j = within & 15;
  int nt = tile / KT_HH, kt = tile % KT_HH;
  int n = nt * 16 + (lane & 15);                     // B[k][n] = Whh[n][k]
  int k = kt * 32 + kmap(lane, j);
  float v = W[(size_t)n * DD + k] * (M[(size_t)n * DD + k] ? 1.0f : 0.0f);
  out[idx] = (__bf16)v;
}

__global__ void pack_wih(const float* __restrict__ W, __bf16* __restrict__ out) {
  // rows n = 0..127 of W_ih (D x F); M_in is all-true there, zero elsewhere
  int idx = blockIdx.x * 256 + threadIdx.x;
  int tile = idx >> 9, within = idx & 511;
  int lane = within >> 4, j = within & 15;
  int nt = tile / 16, kt = tile % 16;                // 8 n-tiles x 16 k-tiles
  int n = nt * 16 + (lane & 15);
  int k = kt * 32 + kmap(lane, j);
  out[idx] = (__bf16)W[(size_t)n * FF + k];
}

__global__ void pack_wop(const float* __restrict__ W, __bf16* __restrict__ out) {
  // W_op is (F=512, 320); pred[.,f] = sum_h op[h]*W_op[f][h] -> B[k=h][n=f]
  int idx = blockIdx.x * 256 + threadIdx.x;
  int tile = idx >> 9, within = idx & 511;
  int lane = within >> 4, j = within & 15;
  int nt = tile / 10, kt = tile % 10;                // 32 n-tiles x 10 k-tiles
  int f = nt * 16 + (lane & 15);
  int k = kt * 32 + kmap(lane, j);
  out[idx] = (__bf16)W[(size_t)f * S1 + k];
}

__global__ void cvt_bf16(const float* __restrict__ src, __bf16* __restrict__ dst, int n) {
  int i = blockIdx.x * 256 + threadIdx.x;
  if (i < n) dst[i] = (__bf16)src[i];
}

__global__ void init_sync(unsigned* s) {
  if (threadIdx.x < 2) s[threadIdx.x] = 0;
}

// ---------------------------------------------------------------------------
// K1: u[r][n] = (x @ Wih^T)[r][n], r = b*T+t, n < 128.  fp32 x converted to
// bf16 on the fly; one wave per (m_tile, n_tile), K = 512 (16 k-tiles).
// ---------------------------------------------------------------------------
__global__ void __launch_bounds__(256) gemm_in(const float* __restrict__ x,
                                               const __bf16* __restrict__ wih_pk,
                                               float* __restrict__ u) {
  int lane = threadIdx.x & 31, wave = threadIdx.x >> 5;
  int mt = blockIdx.x;            // 0..2047
  int nt = wave;                  // 0..7
  int hi8 = (lane & 16) ? 8 : 0;
  int m0 = mt * 16 + (lane & 15);
  v8f acc = {};
  for (int kt = 0; kt < 16; ++kt) {
    const float* ap = x + (size_t)m0 * FF + kt * 32 + hi8;
    v4f a0 = *(const v4f*)(ap);
    v4f a1 = *(const v4f*)(ap + 4);
    v4f a2 = *(const v4f*)(ap + 16);
    v4f a3 = *(const v4f*)(ap + 20);
    BF16x16 A;
#pragma unroll
    for (int j = 0; j < 4; ++j) {
      A.h[0][j]     = (__bf16)a0[j];
      A.h[0][j + 4] = (__bf16)a1[j];
      A.h[1][j]     = (__bf16)a2[j];
      A.h[1][j + 4] = (__bf16)a3[j];
    }
    BF16x16 Bf;
    const v8bf* bp = (const v8bf*)(wih_pk + ((size_t)(nt * 16 + kt) * 512) + lane * 16);
    Bf.h[0] = bp[0];
    Bf.h[1] = bp[1];
    acc = __builtin_amdgcn_wmma_f32_16x16x32_bf16(false, A.v, false, Bf.v,
                                                  (short)0, acc, false, false);
  }
  int n = nt * 16 + (lane & 15);
  int mbase = mt * 16 + hi8;
#pragma unroll
  for (int i = 0; i < 8; ++i)
    u[(size_t)(mbase + i) * S0 + n] = acc[i];
}

// ---------------------------------------------------------------------------
// K2: persistent RNN scan.  44 blocks x 8 waves.  Each block owns 2 n-tiles;
// 4 waves split K (11 k-tiles each) per n-tile and reduce through LDS.
// Per step: stage h (bf16) into LDS, WMMA, reduce, bias+u+tanh, write h
// (double-buffered) and the L23 tap, grid barrier.
// ---------------------------------------------------------------------------
__global__ void __launch_bounds__(256) rnn_scan(const __bf16* __restrict__ whh_pk,
                                                const float* __restrict__ u,
                                                const float* __restrict__ b_ih,
                                                const float* __restrict__ b_hh,
                                                __bf16* __restrict__ hbuf,
                                                __bf16* __restrict__ hs,
                                                unsigned* __restrict__ sync) {
  extern __shared__ char smem_raw[];
  __bf16* hls = (__bf16*)smem_raw;                        // [64][LDS_ROW]
  float*  red = (float*)(smem_raw + BB * LDS_ROW * 2);    // [8 waves][32 lanes][32]

  int tid = threadIdx.x, lane = tid & 31, wave = tid >> 5;
  int n_local = wave >> 2, kq = wave & 3;
  int nt = blockIdx.x * 2 + n_local;                      // 0..87
  int n  = nt * 16 + (lane & 15);
  int hi8 = (lane & 16) ? 8 : 0;
  float bias = b_ih[n] + b_hh[n];                         // b_ih applies to all d

  for (int t = 0; t < TT; ++t) {
    const __bf16* hsrc = hbuf + (size_t)(t & 1) * (BB * DD);
    __bf16*       hdst = hbuf + (size_t)((t + 1) & 1) * (BB * DD);

    // stage full h into LDS with padded stride (all 256 threads, b128 copies)
    for (int c = tid; c < BB * (DD / 8); c += 256) {
      int row = c / (DD / 8), col = c % (DD / 8);
      *(v8bf*)(hls + row * LDS_ROW + col * 8) =
          *(const v8bf*)(hsrc + (size_t)row * DD + col * 8);
    }
    __syncthreads();

    v8f acc[4] = {};                                      // 4 m-tiles (64 rows)
    for (int kk = 0; kk < 11; ++kk) {
      int kt = kq * 11 + kk;
      BF16x16 Bf;
      const v8bf* bp =
          (const v8bf*)(whh_pk + ((size_t)(nt * KT_HH + kt) * 512) + lane * 16);
      Bf.h[0] = bp[0];
      Bf.h[1] = bp[1];
      int col = kt * 32 + hi8;
#pragma unroll
      for (int mt = 0; mt < 4; ++mt) {
        const __bf16* ap = hls + (mt * 16 + (lane & 15)) * LDS_ROW + col;
        BF16x16 A;
        A.h[0] = *(const v8bf*)ap;
        A.h[1] = *(const v8bf*)(ap + 16);
        acc[mt] = __builtin_amdgcn_wmma_f32_16x16x32_bf16(false, A.v, false, Bf.v,
                                                          (short)0, acc[mt], false, false);
      }
    }

    // K-split reduction across the 4 waves sharing this n-tile
    float* myred = red + ((size_t)(wave * 32 + lane) * 32);
#pragma unroll
    for (int mt = 0; mt < 4; ++mt)
#pragma unroll
      for (int i = 0; i < 8; ++i) myred[mt * 8 + i] = acc[mt][i];
    __syncthreads();

    if (kq == 0) {
      const float* r0 = red + ((size_t)(wave * 32 + lane) * 32);  // wave = n_local*4
#pragma unroll
      for (int mt = 0; mt < 4; ++mt) {
#pragma unroll
        for (int i = 0; i < 8; ++i) {
          float s = r0[mt * 8 + i] + r0[1024 + mt * 8 + i] +
                    r0[2048 + mt * 8 + i] + r0[3072 + mt * 8 + i];
          int b = mt * 16 + i + hi8;                      // batch row
          float pre = s + bias;
          if (n < S0) pre += u[((size_t)b * TT + t) * S0 + n];
          float hv = tanhf(pre);
          __bf16 hb = (__bf16)hv;
          hdst[(size_t)b * DD + n] = hb;
          if (n >= S0 && n < S0 + S1)
            hs[((size_t)t * BB + b) * S1 + (n - S0)] = hb;
        }
      }
    }

    // grid barrier across all RNN_BLOCKS workgroups
    __threadfence();
    __syncthreads();
    if (tid == 0) {
      unsigned gen = __hip_atomic_load(&sync[1], __ATOMIC_ACQUIRE, __HIP_MEMORY_SCOPE_AGENT);
      unsigned arr = __hip_atomic_fetch_add(&sync[0], 1u, __ATOMIC_ACQ_REL, __HIP_MEMORY_SCOPE_AGENT);
      if (arr == RNN_BLOCKS - 1) {
        __hip_atomic_store(&sync[0], 0u, __ATOMIC_RELAXED, __HIP_MEMORY_SCOPE_AGENT);
        __hip_atomic_fetch_add(&sync[1], 1u, __ATOMIC_RELEASE, __HIP_MEMORY_SCOPE_AGENT);
      } else {
        while (__hip_atomic_load(&sync[1], __ATOMIC_ACQUIRE, __HIP_MEMORY_SCOPE_AGENT) == gen)
          __builtin_amdgcn_s_sleep(1);
      }
    }
    __syncthreads();
    __threadfence();
  }
}

// ---------------------------------------------------------------------------
// K3: pred[b][t][f] = hs[t][b][:] @ W_op[f][:] + b_op[f]  (rows r = t*64+b,
// which is exactly hs's layout); fused (t,b,f)->(b,t,f) transpose on store.
// ---------------------------------------------------------------------------
__global__ void __launch_bounds__(256) gemm_out(const __bf16* __restrict__ hs,
                                                const __bf16* __restrict__ wop_pk,
                                                const float* __restrict__ b_op,
                                                float* __restrict__ out) {
  int lane = threadIdx.x & 31, wave = threadIdx.x >> 5;
  int mt = blockIdx.x >> 2;                 // 0..2047
  int nt = (blockIdx.x & 3) * 8 + wave;     // 0..31
  int hi8 = (lane & 16) ? 8 : 0;
  int m0 = mt * 16 + (lane & 15);
  v8f acc = {};
  for (int kt = 0; kt < 10; ++kt) {
    const __bf16* ap = hs + (size_t)m0 * S1 + kt * 32 + hi8;
    BF16x16 A;
    A.h[0] = *(const v8bf*)ap;
    A.h[1] = *(const v8bf*)(ap + 16);
    BF16x16 Bf;
    const v8bf* bp = (const v8bf*)(wop_pk + ((size_t)(nt * 10 + kt) * 512) + lane * 16);
    Bf.h[0] = bp[0];
    Bf.h[1] = bp[1];
    acc = __builtin_amdgcn_wmma_f32_16x16x32_bf16(false, A.v, false, Bf.v,
                                                  (short)0, acc, false, false);
  }
  int f = nt * 16 + (lane & 15);
  float bo = b_op[f];
#pragma unroll
  for (int i = 0; i < 8; ++i) {
    int m = mt * 16 + i + hi8;              // = t*64 + b
    int t = m >> 6, b = m & 63;
    out[((size_t)b * TT + t) * FF + f] = acc[i] + bo;
  }
}

// ---------------------------------------------------------------------------
extern "C" void kernel_launch(void* const* d_in, const int* in_sizes, int n_in,
                              void* d_out, int out_size, void* d_ws, size_t ws_size,
                              hipStream_t stream) {
  const float* x    = (const float*)d_in[0];   // (64,512,512)
  const float* h0   = (const float*)d_in[1];   // (64,1408)
  const float* W_ih = (const float*)d_in[2];   // (1408,512)
  const float* W_hh = (const float*)d_in[3];   // (1408,1408)
  const float* b_ih = (const float*)d_in[4];
  const float* b_hh = (const float*)d_in[5];
  const float* W_op = (const float*)d_in[6];   // (512,320)
  const float* b_op = (const float*)d_in[7];
  // d_in[8] = M_in (unused: rows<128 are all-true, rest zero — handled structurally)
  const unsigned char* M_hh = (const unsigned char*)d_in[9];  // bool, 1 byte/elem
  float* out = (float*)d_out;

  // Workspace partition (≈42.5 MB total), 256B aligned
  char* ws = (char*)d_ws;
  size_t off = 0;
  auto alloc = [&](size_t bytes) {
    char* p = ws + off;
    off = (off + bytes + 255) & ~(size_t)255;
    return p;
  };
  __bf16*   whh_pk = (__bf16*)alloc((size_t)NT_HH * KT_HH * 512 * 2);  // 3.96 MB
  __bf16*   wih_pk = (__bf16*)alloc((size_t)8 * 16 * 512 * 2);         // 128 KB
  __bf16*   wop_pk = (__bf16*)alloc((size_t)32 * 10 * 512 * 2);        // 320 KB
  float*    u      = (float*)alloc((size_t)BB * TT * S0 * 4);          // 16 MB
  __bf16*   hsbuf  = (__bf16*)alloc((size_t)TT * BB * S1 * 2);         // 20 MB
  __bf16*   hbuf   = (__bf16*)alloc((size_t)2 * BB * DD * 2);          // 352 KB
  unsigned* sync   = (unsigned*)alloc(256);

  // Phase 0: mask + pack weights, convert h0, reset barrier
  pack_whh<<<(NT_HH * KT_HH * 512) / 256, 256, 0, stream>>>(W_hh, M_hh, whh_pk);
  pack_wih<<<(8 * 16 * 512) / 256, 256, 0, stream>>>(W_ih, wih_pk);
  pack_wop<<<(32 * 10 * 512) / 256, 256, 0, stream>>>(W_op, wop_pk);
  cvt_bf16<<<(BB * DD + 255) / 256, 256, 0, stream>>>(h0, hbuf, BB * DD);
  init_sync<<<1, 32, 0, stream>>>(sync);

  // K1: input projection (only the 128 live columns)
  gemm_in<<<(BB * TT) / 16, 256, 0, stream>>>(x, wih_pk, u);

  // K2: persistent sequential scan with grid barrier
  rnn_scan<<<RNN_BLOCKS, 256, RNN_LDS_BYTES, stream>>>(whh_pk, u, b_ih, b_hh,
                                                       hbuf, hsbuf, sync);

  // K3: output projection + transpose + bias
  gemm_out<<<(BB * TT / 16) * 4, 256, 0, stream>>>(hsbuf, wop_pk, b_op, out);
}
